// GNS_18408229831062
// MI455X (gfx1250) — matrix operand order repaired and checked by reading, not compiled, not run on hardware
//
#include <hip/hip_runtime.h>

// ---------------------------------------------------------------------------
// GNS forward for MI455X (gfx1250, wave32).  f16 WMMA GEMMs, f32 accumulate,
// fused MLP+LN+residual per kernel, f32 global atomics for segment_sum.
// ---------------------------------------------------------------------------

typedef __attribute__((ext_vector_type(16))) _Float16 v16h;
typedef __attribute__((ext_vector_type(8)))  float    v8f;

#define NN 20000
#define EEC 160000
#define HD 128
#define LLC 10

static __device__ __forceinline__ v8f wmma16(v16h a, v16h b, v8f c) {
  return __builtin_amdgcn_wmma_f32_16x16x32_f16(false, a, false, b, (short)0, c, false, false);
}

static __device__ __forceinline__ void zero_acc(v8f acc[8]) {
#pragma unroll
  for (int t = 0; t < 8; ++t)
#pragma unroll
    for (int i = 0; i < 8; ++i) acc[t][i] = 0.f;
}

// A fragment (16x32 f16) built from a per-lane f32 row pointer.
// Lane holds row m=lane&15; K = e + (lane>=16)*16  (contiguous, matches B).
static __device__ __forceinline__ v16h afrag_f32(const float* __restrict__ p, int klo) {
  const float4 f0 = *(const float4*)(p + 2 * klo);
  const float4 f1 = *(const float4*)(p + 2 * klo + 4);
  const float4 f2 = *(const float4*)(p + 2 * klo + 8);
  const float4 f3 = *(const float4*)(p + 2 * klo + 12);
  v16h a;
  a[0]=(_Float16)f0.x; a[1]=(_Float16)f0.y; a[2]=(_Float16)f0.z; a[3]=(_Float16)f0.w;
  a[4]=(_Float16)f1.x; a[5]=(_Float16)f1.y; a[6]=(_Float16)f1.z; a[7]=(_Float16)f1.w;
  a[8]=(_Float16)f2.x; a[9]=(_Float16)f2.y; a[10]=(_Float16)f2.z; a[11]=(_Float16)f2.w;
  a[12]=(_Float16)f3.x; a[13]=(_Float16)f3.y; a[14]=(_Float16)f3.z; a[15]=(_Float16)f3.w;
  return a;
}

// A fragment from wave-private f16 LDS row (row start pointer + k-window base).
static __device__ __forceinline__ v16h afrag_lds(const _Float16* __restrict__ p, int klo) {
  v16h a;
#pragma unroll
  for (int i = 0; i < 8; ++i) { a[i] = p[2 * klo + i]; a[i + 8] = p[2 * klo + 8 + i]; }
  return a;
}

// GEMM: 16 x KP (f16 in LDS, row stride KP)  x  Wt[128][KP]  -> acc[8] (16x128)
template <int KP>
static __device__ __forceinline__ void gemm_ldsA(const _Float16* __restrict__ lA,
                                                 const _Float16* __restrict__ Wt,
                                                 v8f acc[8], int lane) {
  const int m = lane & 15, klo = (lane >> 4) * 8, kbB = (lane >> 4) * 16, n = lane & 15;
#pragma unroll
  for (int kt = 0; kt < KP / 32; ++kt) {
    v16h a = afrag_lds(lA + m * KP + kt * 32, klo);
#pragma unroll
    for (int nt = 0; nt < 8; ++nt) {
      v16h b = *(const v16h*)(Wt + (size_t)(nt * 16 + n) * KP + kt * 32 + kbB);
      acc[nt] = wmma16(a, b, acc[nt]);
    }
  }
}

// Add bias, ReLU, write 16x128 f16 tile to wave-private LDS (C layout ISA-defined:
// lane n=lane&15, rows rowoff+i for i=0..7, rowoff=(lane>=16)*8).
static __device__ __forceinline__ void store_act(const v8f acc[8], const float* __restrict__ bias,
                                                 _Float16* __restrict__ lh, int lane) {
  const int n = lane & 15, rowoff = (lane >> 4) * 8;
#pragma unroll
  for (int nt = 0; nt < 8; ++nt) {
    const float bn = bias[nt * 16 + n];
#pragma unroll
    for (int i = 0; i < 8; ++i) {
      float v = acc[nt][i] + bn;
      v = fmaxf(v, 0.f);
      lh[(rowoff + i) * HD + nt * 16 + n] = (_Float16)v;
    }
  }
}

static __device__ __forceinline__ void add_bias(v8f acc[8], const float* __restrict__ bias, int lane) {
  const int n = lane & 15;
#pragma unroll
  for (int nt = 0; nt < 8; ++nt) {
    const float bn = bias[nt * 16 + n];
#pragma unroll
    for (int i = 0; i < 8; ++i) acc[nt][i] += bn;
  }
}

// In-register LayerNorm stats: per-row sum/sumsq across 8 n-tiles + 16-lane xor-reduce.
static __device__ __forceinline__ void ln_stats(const v8f acc[8], float mu[8], float rs[8]) {
#pragma unroll
  for (int i = 0; i < 8; ++i) {
    float s = 0.f, q = 0.f;
#pragma unroll
    for (int nt = 0; nt < 8; ++nt) { const float v = acc[nt][i]; s += v; q += v * v; }
    for (int d = 1; d < 16; d <<= 1) { s += __shfl_xor(s, d, 32); q += __shfl_xor(q, d, 32); }
    const float m = s * (1.f / 128.f);
    const float var = q * (1.f / 128.f) - m * m;
    mu[i] = m;
    rs[i] = rsqrtf(var + 1e-5f);
  }
}

// ---------------------------------------------------------------------------
// Weight prep: W[b][K][Hn] (f32) -> Wt[b][Hn][KP] (f16), zero-padded k>=K.
// ---------------------------------------------------------------------------
__global__ __launch_bounds__(256) void k_wprep(const float* __restrict__ W, _Float16* __restrict__ Wt,
                                               int K, int Hn, int KP, long total) {
  const long stride = (long)gridDim.x * blockDim.x;
  for (long idx = (long)blockIdx.x * blockDim.x + threadIdx.x; idx < total; idx += stride) {
    const long per = (long)Hn * KP;
    const long b = idx / per;
    const long rem = idx - b * per;
    const int nI = (int)(rem / KP);
    const int k = (int)(rem - (long)nI * KP);
    const float v = (k < K) ? W[(b * K + k) * (long)Hn + nI] : 0.f;
    Wt[idx] = (_Float16)v;
  }
}

// ---------------------------------------------------------------------------
// Node encoder: features (37 -> pad 64) -> 128 -> 128 -> 128, LN.
// ---------------------------------------------------------------------------
__global__ __launch_bounds__(256) void k_node_encode(
    const int* __restrict__ mats, const float* __restrict__ vel, const float* __restrict__ pos,
    const float* __restrict__ matW, const float* __restrict__ matB,
    const _Float16* __restrict__ Wt0, const float* __restrict__ b0,
    const _Float16* __restrict__ Wt1, const float* __restrict__ b1,
    const _Float16* __restrict__ Wt2, const float* __restrict__ b2,
    const float* __restrict__ g, const float* __restrict__ lb,
    float* __restrict__ nodes) {
  const int wave = threadIdx.x >> 5, lane = threadIdx.x & 31;
  const int base = (blockIdx.x * 8 + wave) * 16;
  if (base >= NN) return;
  __shared__ _Float16 shALL[8][16 * HD];
  _Float16* lh = shALL[wave];

  // Stage features 16x64 (stride 64), zero-padded beyond col 37.
  for (int idx = lane; idx < 16 * 64; idx += 32) {
    const int row = idx >> 6, c = idx & 63;
    const int i = base + row;
    float v = 0.f;
    if (c < 15) {
      v = vel[(size_t)i * 15 + c];
    } else if (c < 31) {
      const int j = c - 15;
      v = matW[mats[i] * 16 + j] + matB[j];
    } else if (c < 37) {
      const int j = c - 31, d = j >> 1;
      const float pv = pos[(size_t)i * 3 + d];
      const float w = (j & 1) ? (1.0f - pv) : pv;
      v = fminf(fmaxf(w, 0.f), 0.03f);
    }
    lh[idx] = (_Float16)v;
  }
  v8f acc[8];
  zero_acc(acc);
  gemm_ldsA<64>(lh, Wt0, acc, lane);
  store_act(acc, b0, lh, lane);
  zero_acc(acc);
  gemm_ldsA<128>(lh, Wt1, acc, lane);
  store_act(acc, b1, lh, lane);
  zero_acc(acc);
  gemm_ldsA<128>(lh, Wt2, acc, lane);
  add_bias(acc, b2, lane);
  float mu[8], rs[8];
  ln_stats(acc, mu, rs);
  const int n = lane & 15, rowoff = (lane >> 4) * 8;
#pragma unroll
  for (int nt = 0; nt < 8; ++nt) {
    const int c = nt * 16 + n;
    const float gc = g[c], lc = lb[c];
#pragma unroll
    for (int i = 0; i < 8; ++i)
      nodes[(size_t)(base + rowoff + i) * HD + c] = (acc[nt][i] - mu[i]) * rs[i] * gc + lc;
  }
}

// ---------------------------------------------------------------------------
// Edge encoder: [rel(3), dist] (pad 32) -> 128 -> 128 -> 128, LN.
// ---------------------------------------------------------------------------
__global__ __launch_bounds__(256) void k_edge_encode(
    const float* __restrict__ pos, const int* __restrict__ nbr,
    const _Float16* __restrict__ Wt0, const float* __restrict__ b0,
    const _Float16* __restrict__ Wt1, const float* __restrict__ b1,
    const _Float16* __restrict__ Wt2, const float* __restrict__ b2,
    const float* __restrict__ g, const float* __restrict__ lb,
    float* __restrict__ edges) {
  const int wave = threadIdx.x >> 5, lane = threadIdx.x & 31;
  const int ebase = (blockIdx.x * 8 + wave) * 16;
  if (ebase >= EEC) return;
  __shared__ _Float16 shALL[8][16 * HD];
  _Float16* lh = shALL[wave];

  for (int idx = lane; idx < 16 * 32; idx += 32) {
    const int row = idx >> 5, c = idx & 31;
    float v = 0.f;
    if (c < 4) {
      const int e = ebase + row;
      const int r = nbr[2 * e], s = nbr[2 * e + 1];
      const float rx = pos[(size_t)r * 3 + 0] - pos[(size_t)s * 3 + 0];
      const float ry = pos[(size_t)r * 3 + 1] - pos[(size_t)s * 3 + 1];
      const float rz = pos[(size_t)r * 3 + 2] - pos[(size_t)s * 3 + 2];
      v = (c == 0) ? rx : (c == 1) ? ry : (c == 2) ? rz : sqrtf(rx * rx + ry * ry + rz * rz);
    }
    lh[idx] = (_Float16)v;
  }
  v8f acc[8];
  zero_acc(acc);
  gemm_ldsA<32>(lh, Wt0, acc, lane);
  store_act(acc, b0, lh, lane);
  zero_acc(acc);
  gemm_ldsA<128>(lh, Wt1, acc, lane);
  store_act(acc, b1, lh, lane);
  zero_acc(acc);
  gemm_ldsA<128>(lh, Wt2, acc, lane);
  add_bias(acc, b2, lane);
  float mu[8], rs[8];
  ln_stats(acc, mu, rs);
  const int n = lane & 15, rowoff = (lane >> 4) * 8;
#pragma unroll
  for (int nt = 0; nt < 8; ++nt) {
    const int c = nt * 16 + n;
    const float gc = g[c], lc = lb[c];
#pragma unroll
    for (int i = 0; i < 8; ++i)
      edges[(size_t)(ebase + rowoff + i) * HD + c] = (acc[nt][i] - mu[i]) * rs[i] * gc + lc;
  }
}

// ---------------------------------------------------------------------------
// Edge processor layer: MLP([e, n_r, n_s]) K=384 -> 128 -> 128, LN, residual,
// and fused segment-sum (f32 atomics into agg[ridx]).
// ---------------------------------------------------------------------------
__global__ __launch_bounds__(256) void k_edge_layer(
    float* __restrict__ edges, const float* __restrict__ nodes, float* __restrict__ agg,
    const int* __restrict__ nbr,
    const _Float16* __restrict__ Wt0, const float* __restrict__ b0,
    const _Float16* __restrict__ Wt1, const float* __restrict__ b1,
    const _Float16* __restrict__ Wt2, const float* __restrict__ b2,
    const float* __restrict__ g, const float* __restrict__ lb) {
  const int wave = threadIdx.x >> 5, lane = threadIdx.x & 31;
  const int ebase = (blockIdx.x * 8 + wave) * 16;
  if (ebase >= EEC) return;
  __shared__ _Float16 shALL[8][16 * HD];
  _Float16* lh = shALL[wave];
  const int m = lane & 15, klo = (lane >> 4) * 8, kbB = (lane >> 4) * 16, n = lane & 15;
  const int erow = ebase + m;
  const int r = nbr[2 * erow], s = nbr[2 * erow + 1];
  const float* p0 = edges + (size_t)erow * HD;
  const float* p1 = nodes + (size_t)r * HD;
  const float* p2 = nodes + (size_t)s * HD;

  v8f acc[8];
  zero_acc(acc);
#pragma unroll
  for (int blk = 0; blk < 3; ++blk) {
    const float* p = (blk == 0) ? p0 : (blk == 1) ? p1 : p2;
    for (int kt = 0; kt < 4; ++kt) {
      v16h a = afrag_f32(p + kt * 32, klo);
#pragma unroll
      for (int nt = 0; nt < 8; ++nt) {
        v16h b = *(const v16h*)(Wt0 + (size_t)(nt * 16 + n) * 384 + (blk * 4 + kt) * 32 + kbB);
        acc[nt] = wmma16(a, b, acc[nt]);
      }
    }
  }
  store_act(acc, b0, lh, lane);
  zero_acc(acc);
  gemm_ldsA<128>(lh, Wt1, acc, lane);
  store_act(acc, b1, lh, lane);
  zero_acc(acc);
  gemm_ldsA<128>(lh, Wt2, acc, lane);
  add_bias(acc, b2, lane);
  float mu[8], rs[8];
  ln_stats(acc, mu, rs);
  const int rowoff = (lane >> 4) * 8;
  int rr[8];
#pragma unroll
  for (int i = 0; i < 8; ++i) rr[i] = nbr[2 * (ebase + rowoff + i)];
#pragma unroll
  for (int nt = 0; nt < 8; ++nt) {
    const int c = nt * 16 + n;
    const float gc = g[c], lc = lb[c];
#pragma unroll
    for (int i = 0; i < 8; ++i) {
      const size_t off = (size_t)(ebase + rowoff + i) * HD + c;
      const float v = edges[off] + (acc[nt][i] - mu[i]) * rs[i] * gc + lc;
      edges[off] = v;
      atomicAdd(&agg[(size_t)rr[i] * HD + c], v);
    }
  }
}

// ---------------------------------------------------------------------------
// Node processor layer: MLP([agg, node]) K=256 -> 128 -> 128, LN, residual.
// ---------------------------------------------------------------------------
__global__ __launch_bounds__(256) void k_node_layer(
    float* __restrict__ nodes, const float* __restrict__ agg,
    const _Float16* __restrict__ Wt0, const float* __restrict__ b0,
    const _Float16* __restrict__ Wt1, const float* __restrict__ b1,
    const _Float16* __restrict__ Wt2, const float* __restrict__ b2,
    const float* __restrict__ g, const float* __restrict__ lb) {
  const int wave = threadIdx.x >> 5, lane = threadIdx.x & 31;
  const int base = (blockIdx.x * 8 + wave) * 16;
  if (base >= NN) return;
  __shared__ _Float16 shALL[8][16 * HD];
  _Float16* lh = shALL[wave];
  const int m = lane & 15, klo = (lane >> 4) * 8, kbB = (lane >> 4) * 16, n = lane & 15;
  const int row = base + m;
  const float* p0 = agg + (size_t)row * HD;
  const float* p1 = nodes + (size_t)row * HD;

  v8f acc[8];
  zero_acc(acc);
#pragma unroll
  for (int blk = 0; blk < 2; ++blk) {
    const float* p = (blk == 0) ? p0 : p1;
    for (int kt = 0; kt < 4; ++kt) {
      v16h a = afrag_f32(p + kt * 32, klo);
#pragma unroll
      for (int nt = 0; nt < 8; ++nt) {
        v16h b = *(const v16h*)(Wt0 + (size_t)(nt * 16 + n) * 256 + (blk * 4 + kt) * 32 + kbB);
        acc[nt] = wmma16(a, b, acc[nt]);
      }
    }
  }
  store_act(acc, b0, lh, lane);
  zero_acc(acc);
  gemm_ldsA<128>(lh, Wt1, acc, lane);
  store_act(acc, b1, lh, lane);
  zero_acc(acc);
  gemm_ldsA<128>(lh, Wt2, acc, lane);
  add_bias(acc, b2, lane);
  float mu[8], rs[8];
  ln_stats(acc, mu, rs);
  const int rowoff = (lane >> 4) * 8;
#pragma unroll
  for (int nt = 0; nt < 8; ++nt) {
    const int c = nt * 16 + n;
    const float gc = g[c], lc = lb[c];
#pragma unroll
    for (int i = 0; i < 8; ++i) {
      const size_t off = (size_t)(base + rowoff + i) * HD + c;
      nodes[off] = nodes[off] + (acc[nt][i] - mu[i]) * rs[i] * gc + lc;
    }
  }
}

// ---------------------------------------------------------------------------
// Decoder: 128 -> 128 -> 128 (WMMA, ReLU) -> 3 (VALU).
// ---------------------------------------------------------------------------
__global__ __launch_bounds__(256) void k_decode(
    const float* __restrict__ nodes,
    const _Float16* __restrict__ Wt0, const float* __restrict__ b0,
    const _Float16* __restrict__ Wt1, const float* __restrict__ b1,
    const float* __restrict__ W2, const float* __restrict__ b2,
    float* __restrict__ out) {
  const int wave = threadIdx.x >> 5, lane = threadIdx.x & 31;
  const int base = (blockIdx.x * 8 + wave) * 16;
  if (base >= NN) return;
  __shared__ _Float16 shALL[8][16 * HD];
  _Float16* lh = shALL[wave];
  const int klo = (lane >> 4) * 8, kbB = (lane >> 4) * 16, n = lane & 15, m = lane & 15;
  const float* p = nodes + (size_t)(base + m) * HD;

  v8f acc[8];
  zero_acc(acc);
  for (int kt = 0; kt < 4; ++kt) {
    v16h a = afrag_f32(p + kt * 32, klo);
#pragma unroll
    for (int nt = 0; nt < 8; ++nt) {
      v16h b = *(const v16h*)(Wt0 + (size_t)(nt * 16 + n) * HD + kt * 32 + kbB);
      acc[nt] = wmma16(a, b, acc[nt]);
    }
  }
  store_act(acc, b0, lh, lane);
  zero_acc(acc);
  gemm_ldsA<128>(lh, Wt1, acc, lane);
  store_act(acc, b1, lh, lane);
  // Final 128 -> 3 with VALU from LDS f16 (h2) and f32 W2 (128x3).
#pragma unroll
  for (int t = 0; t < 2; ++t) {
    const int id = t * 32 + lane;
    if (id < 48) {
      const int j = id >> 4, mm = id & 15;
      float s = b2[j];
      for (int k = 0; k < HD; ++k) s += (float)lh[mm * HD + k] * W2[k * 3 + j];
      out[(size_t)(base + mm) * 3 + j] = s;
    }
  }
}

// ---------------------------------------------------------------------------
// Host side
// ---------------------------------------------------------------------------
extern "C" void kernel_launch(void* const* d_in, const int* in_sizes, int n_in,
                              void* d_out, int out_size, void* d_ws, size_t ws_size,
                              hipStream_t stream) {
  (void)in_sizes; (void)n_in; (void)out_size; (void)ws_size;
  // ---- inputs (setup_inputs dict order, nested dict depth-first) ----
  const int*   mats  = (const int*)d_in[0];
  const float* vel   = (const float*)d_in[1];
  const float* pos   = (const float*)d_in[2];
  const int*   nbr   = (const int*)d_in[3];
  const float* mat_W = (const float*)d_in[4];
  const float* mat_b = (const float*)d_in[5];
  const float* ne_W0 = (const float*)d_in[6],  *ne_b0 = (const float*)d_in[7];
  const float* ne_W1 = (const float*)d_in[8],  *ne_b1 = (const float*)d_in[9];
  const float* ne_W2 = (const float*)d_in[10], *ne_b2 = (const float*)d_in[11];
  const float* ne_g  = (const float*)d_in[12], *ne_lb = (const float*)d_in[13];
  const float* ee_W0 = (const float*)d_in[14], *ee_b0 = (const float*)d_in[15];
  const float* ee_W1 = (const float*)d_in[16], *ee_b1 = (const float*)d_in[17];
  const float* ee_W2 = (const float*)d_in[18], *ee_b2 = (const float*)d_in[19];
  const float* ee_g  = (const float*)d_in[20], *ee_lb = (const float*)d_in[21];
  const float* pe_W0 = (const float*)d_in[22], *pe_b0 = (const float*)d_in[23];
  const float* pe_W1 = (const float*)d_in[24], *pe_b1 = (const float*)d_in[25];
  const float* pe_W2 = (const float*)d_in[26], *pe_b2 = (const float*)d_in[27];
  const float* pe_g  = (const float*)d_in[28], *pe_lb = (const float*)d_in[29];
  const float* pn_W0 = (const float*)d_in[30], *pn_b0 = (const float*)d_in[31];
  const float* pn_W1 = (const float*)d_in[32], *pn_b1 = (const float*)d_in[33];
  const float* pn_W2 = (const float*)d_in[34], *pn_b2 = (const float*)d_in[35];
  const float* pn_g  = (const float*)d_in[36], *pn_lb = (const float*)d_in[37];
  const float* dc_W0 = (const float*)d_in[38], *dc_b0 = (const float*)d_in[39];
  const float* dc_W1 = (const float*)d_in[40], *dc_b1 = (const float*)d_in[41];
  const float* dc_W2 = (const float*)d_in[42], *dc_b2 = (const float*)d_in[43];

  // ---- workspace carve-up (all chunks are 256B-aligned sizes) ----
  char* W = (char*)d_ws;
  size_t o = 0;
  float* nodes = (float*)(W + o); o += (size_t)NN * HD * 4;
  float* edges = (float*)(W + o); o += (size_t)EEC * HD * 4;
  float* agg   = (float*)(W + o); o += (size_t)NN * HD * 4;
  _Float16* ne0t = (_Float16*)(W + o); o += 128 * 64 * 2;
  _Float16* ne1t = (_Float16*)(W + o); o += 128 * 128 * 2;
  _Float16* ne2t = (_Float16*)(W + o); o += 128 * 128 * 2;
  _Float16* ee0t = (_Float16*)(W + o); o += 128 * 32 * 2;
  _Float16* ee1t = (_Float16*)(W + o); o += 128 * 128 * 2;
  _Float16* ee2t = (_Float16*)(W + o); o += 128 * 128 * 2;
  _Float16* pe0t = (_Float16*)(W + o); o += (size_t)LLC * 128 * 384 * 2;
  _Float16* pe1t = (_Float16*)(W + o); o += (size_t)LLC * 128 * 128 * 2;
  _Float16* pe2t = (_Float16*)(W + o); o += (size_t)LLC * 128 * 128 * 2;
  _Float16* pn0t = (_Float16*)(W + o); o += (size_t)LLC * 128 * 256 * 2;
  _Float16* pn1t = (_Float16*)(W + o); o += (size_t)LLC * 128 * 128 * 2;
  _Float16* pn2t = (_Float16*)(W + o); o += (size_t)LLC * 128 * 128 * 2;
  _Float16* d0t  = (_Float16*)(W + o); o += 128 * 128 * 2;
  _Float16* d1t  = (_Float16*)(W + o); o += 128 * 128 * 2;

  auto WP = [&](const float* src, _Float16* dst, int K, int Hn, int KP, long B) {
    const long total = B * (long)Hn * KP;
    int blocks = (int)((total + 255) / 256);
    if (blocks > 4096) blocks = 4096;
    k_wprep<<<blocks, 256, 0, stream>>>(src, dst, K, Hn, KP, total);
  };
  WP(ne_W0, ne0t, 37, 128, 64, 1);
  WP(ne_W1, ne1t, 128, 128, 128, 1);
  WP(ne_W2, ne2t, 128, 128, 128, 1);
  WP(ee_W0, ee0t, 4, 128, 32, 1);
  WP(ee_W1, ee1t, 128, 128, 128, 1);
  WP(ee_W2, ee2t, 128, 128, 128, 1);
  WP(pe_W0, pe0t, 384, 128, 384, LLC);
  WP(pe_W1, pe1t, 128, 128, 128, LLC);
  WP(pe_W2, pe2t, 128, 128, 128, LLC);
  WP(pn_W0, pn0t, 256, 128, 256, LLC);
  WP(pn_W1, pn1t, 128, 128, 128, LLC);
  WP(pn_W2, pn2t, 128, 128, 128, LLC);
  WP(dc_W0, d0t, 128, 128, 128, 1);
  WP(dc_W1, d1t, 128, 128, 128, 1);

  const int nodeBlocks = (NN / 16 + 7) / 8;   // 157
  const int edgeBlocks = (EEC / 16) / 8;      // 1250

  k_node_encode<<<nodeBlocks, 256, 0, stream>>>(mats, vel, pos, mat_W, mat_b,
                                                ne0t, ne_b0, ne1t, ne_b1, ne2t, ne_b2,
                                                ne_g, ne_lb, nodes);
  k_edge_encode<<<edgeBlocks, 256, 0, stream>>>(pos, nbr,
                                                ee0t, ee_b0, ee1t, ee_b1, ee2t, ee_b2,
                                                ee_g, ee_lb, edges);

  for (int l = 0; l < LLC; ++l) {
    hipMemsetAsync(agg, 0, (size_t)NN * HD * 4, stream);
    k_edge_layer<<<edgeBlocks, 256, 0, stream>>>(
        edges, nodes, agg, nbr,
        pe0t + (size_t)l * 128 * 384, pe_b0 + (size_t)l * 128,
        pe1t + (size_t)l * 128 * 128, pe_b1 + (size_t)l * 128,
        pe2t + (size_t)l * 128 * 128, pe_b2 + (size_t)l * 128,
        pe_g + (size_t)l * 128, pe_lb + (size_t)l * 128);
    k_node_layer<<<nodeBlocks, 256, 0, stream>>>(
        nodes, agg,
        pn0t + (size_t)l * 128 * 256, pn_b0 + (size_t)l * 128,
        pn1t + (size_t)l * 128 * 128, pn_b1 + (size_t)l * 128,
        pn2t + (size_t)l * 128 * 128, pn_b2 + (size_t)l * 128,
        pn_g + (size_t)l * 128, pn_lb + (size_t)l * 128);
  }

  k_decode<<<nodeBlocks, 256, 0, stream>>>(nodes, d0t, dc_b0, d1t, dc_b1,
                                           dc_W2, dc_b2, (float*)d_out);
}
